// KCVLAKeywordVisionCrossAttention_14181982011508
// MI455X (gfx1250) — matrix operand order repaired and checked by reading, not compiled, not run on hardware
//
#include <hip/hip_runtime.h>
#include <hip/hip_bf16.h>
#include <stdint.h>

// ---------------------------------------------------------------------------
// Problem constants
// ---------------------------------------------------------------------------
#define BB 32
#define SS 2048
#define DD 512
#define HH 8
#define HD 64
#define KW 16
#define S_CHUNK 64
#define CH_STRIDE 520   // 512 + 8 bf16 pad; keeps 16B alignment (1040B rows)

typedef unsigned char u8;
typedef __attribute__((ext_vector_type(16))) __bf16 v16bf;
typedef __attribute__((ext_vector_type(8)))  float  v8f;

union FragU { v16bf v; uint4 q[2]; };

// ---------------------------------------------------------------------------
// bf16 helpers (round-to-nearest-even)
// ---------------------------------------------------------------------------
__device__ __forceinline__ unsigned short f32_to_bf16(float f) {
  unsigned int u = __float_as_uint(f);
  unsigned int r = u + 0x7FFFu + ((u >> 16) & 1u);
  return (unsigned short)(r >> 16);
}
__device__ __forceinline__ float bf16_to_f32(unsigned short h) {
  return __uint_as_float(((unsigned int)h) << 16);
}

// ---------------------------------------------------------------------------
// WMMA fragment loads (wave32, 16x16x32 bf16).
// A (16x32, MxK): lane L holds row M = L%16; K runs [kb,kb+8) and [16+kb,16+kb+8),
//                 kb = 8*(L>=16)  (per ISA 7.12.2 "16-bit A-Matrix 16x32").
// B (32x16, KxN): lane L holds col N = L%16; K runs [kb,kb+8) and [16+kb,16+kb+8), kb = 8*(L>=16).
// C/D (16x16 f32): lane L holds N = L%16; VGPR r holds M = r + 8*(L>=16).
// ---------------------------------------------------------------------------
__device__ __forceinline__ v16bf load_fragA(const unsigned short* base, int row0,
                                            int stride, int k0) {
  int lane = threadIdx.x & 31;
  const unsigned short* p = base + (row0 + (lane & 15)) * stride + k0 + ((lane >> 4) << 3);
  FragU f;
  f.q[0] = *reinterpret_cast<const uint4*>(p);
  f.q[1] = *reinterpret_cast<const uint4*>(p + 16);
  return f.v;
}
__device__ __forceinline__ v16bf load_fragB(const unsigned short* base, int n0,
                                            int stride, int k0) {
  int lane = threadIdx.x & 31;
  const unsigned short* p = base + (n0 + (lane & 15)) * stride + k0 + ((lane >> 4) << 3);
  FragU f;
  f.q[0] = *reinterpret_cast<const uint4*>(p);
  f.q[1] = *reinterpret_cast<const uint4*>(p + 16);
  return f.v;
}
__device__ __forceinline__ v8f wmma_bf16(v16bf a, v16bf b, v8f c) {
  return __builtin_amdgcn_wmma_f32_16x16x32_bf16(false, a, false, b, (short)0, c,
                                                 false, false);
}

// ---------------------------------------------------------------------------
// Block reductions (wave32 shuffles + LDS, 256-thread blocks = 8 waves)
// ---------------------------------------------------------------------------
__device__ __forceinline__ float blockReduceSum(float v, volatile float* sm) {
  for (int o = 16; o; o >>= 1) v += __shfl_xor(v, o);
  int w = threadIdx.x >> 5;
  __syncthreads();
  if ((threadIdx.x & 31) == 0) sm[w] = v;
  __syncthreads();
  if (threadIdx.x == 0) {
    float t = 0.f;
    for (int i = 0; i < 8; ++i) t += sm[i];
    sm[8] = t;
  }
  __syncthreads();
  return sm[8];
}
__device__ __forceinline__ float blockReduceMax(float v, volatile float* sm) {
  for (int o = 16; o; o >>= 1) v = fmaxf(v, __shfl_xor(v, o));
  int w = threadIdx.x >> 5;
  __syncthreads();
  if ((threadIdx.x & 31) == 0) sm[w] = v;
  __syncthreads();
  if (threadIdx.x == 0) {
    float t = -3.402823466e38f;
    for (int i = 0; i < 8; ++i) t = fmaxf(t, sm[i]);
    sm[8] = t;
  }
  __syncthreads();
  return sm[8];
}

// ---------------------------------------------------------------------------
// Kernel W: convert in_proj_w (1536x512 f32) -> bf16 scratch
// ---------------------------------------------------------------------------
__global__ __launch_bounds__(256) void k_convW(const float* __restrict__ w,
                                               unsigned short* __restrict__ wb) {
  int idx4 = blockIdx.x * 256 + threadIdx.x;            // 196608 float4 total
  float4 v = *reinterpret_cast<const float4*>(w + (size_t)idx4 * 4);
  unsigned int lo = (unsigned int)f32_to_bf16(v.x) | ((unsigned int)f32_to_bf16(v.y) << 16);
  unsigned int hi = (unsigned int)f32_to_bf16(v.z) | ((unsigned int)f32_to_bf16(v.w) << 16);
  *reinterpret_cast<uint2*>(wb + (size_t)idx4 * 4) = make_uint2(lo, hi);
}

// ---------------------------------------------------------------------------
// Kernel Q: q = (kq @ Wq^T + bq) * 1/sqrt(HD), stored bf16  (WMMA)
// one block per b; output 16x512, inner 512.
// Wave n-blocking: each wave owns 4 consecutive n-tiles, reuses the A fragment
// across them -> 1 LDS load + 4 global loads + 4 independent WMMAs per k-step.
// ---------------------------------------------------------------------------
__global__ __launch_bounds__(256) void k_qproj(const float* __restrict__ kq,
                                               const unsigned short* __restrict__ Wb,
                                               const float* __restrict__ inb,
                                               unsigned short* __restrict__ qb) {
  __shared__ unsigned short qin[KW * CH_STRIDE];
  int b = blockIdx.x, tid = threadIdx.x;
  // stage keyword queries (16x512 f32) as bf16
  for (int i = 0; i < 8; ++i) {
    int idx4 = tid + i * 256;                            // 2048 float4
    int row = idx4 >> 7, d = (idx4 & 127) << 2;
    float4 v = *reinterpret_cast<const float4*>(kq + ((size_t)(b * KW + row)) * DD + d);
    unsigned int lo = (unsigned int)f32_to_bf16(v.x) | ((unsigned int)f32_to_bf16(v.y) << 16);
    unsigned int hi = (unsigned int)f32_to_bf16(v.z) | ((unsigned int)f32_to_bf16(v.w) << 16);
    *reinterpret_cast<uint2*>(&qin[row * CH_STRIDE + d]) = make_uint2(lo, hi);
  }
  __syncthreads();
  int w = tid >> 5, lane = tid & 31;
  int nt0 = w * 4;                                       // 32 n-tiles / 8 waves
  v8f c[4] = {};
#pragma unroll
  for (int kk = 0; kk < 16; ++kk) {
    v16bf a = load_fragA(qin, 0, CH_STRIDE, kk * 32);
#pragma unroll
    for (int j = 0; j < 4; ++j)
      c[j] = wmma_bf16(a, load_fragB(Wb, (nt0 + j) * 16, DD, kk * 32), c[j]);
  }
#pragma unroll
  for (int j = 0; j < 4; ++j) {
    int n = (nt0 + j) * 16 + (lane & 15);
    float bias = inb[n];
#pragma unroll
    for (int r = 0; r < 8; ++r) {
      int m = r + ((lane >> 4) << 3);
      qb[((size_t)(b * KW + m)) * DD + n] = f32_to_bf16((c[j][r] + bias) * 0.125f);
    }
  }
}

// ---------------------------------------------------------------------------
// Kernel A (dominant): per (b, 64-row s-chunk):
//   stage vision chunk bf16 -> LDS, WMMA-project K (->LDS) and V (->ws bf16),
//   then WMMA q.k^T per head -> masked raw scores (f32 ws).
// M-register-blocking: per n-tile a wave holds 4 C accumulators (one per
// 16-row m-tile); each k-step = 1 global B-fragment load + 4 LDS A loads +
// 4 independent WMMAs, so loadcnt waits amortize over 4 MMAs and weight
// traffic drops 4x vs naive.
// dynamic LDS: chunk(64x520) + kc(64x520) bf16 = 133120 B
// ---------------------------------------------------------------------------
__global__ __launch_bounds__(256) void k_kv_scores(const float* __restrict__ vision,
                                                   const u8* __restrict__ pad,
                                                   const unsigned short* __restrict__ Wb,
                                                   const float* __restrict__ inb,
                                                   const unsigned short* __restrict__ qb,
                                                   unsigned short* __restrict__ Vb,
                                                   float* __restrict__ scores) {
  extern __shared__ unsigned short smem[];
  unsigned short* chunk = smem;                          // [64][CH_STRIDE]
  unsigned short* kc    = smem + S_CHUNK * CH_STRIDE;    // [64][CH_STRIDE]
  int b  = blockIdx.x >> 5;                              // SS / S_CHUNK = 32
  int s0 = (blockIdx.x & 31) * S_CHUNK;
  int tid = threadIdx.x;

  // stage vision chunk (64x512 f32) as bf16
  for (int i = 0; i < 32; ++i) {
    int idx4 = tid + i * 256;                            // 8192 float4
    int sl = idx4 >> 7, d = (idx4 & 127) << 2;
    float4 v = *reinterpret_cast<const float4*>(vision + ((size_t)(b * SS + s0 + sl)) * DD + d);
    unsigned int lo = (unsigned int)f32_to_bf16(v.x) | ((unsigned int)f32_to_bf16(v.y) << 16);
    unsigned int hi = (unsigned int)f32_to_bf16(v.z) | ((unsigned int)f32_to_bf16(v.w) << 16);
    *reinterpret_cast<uint2*>(&chunk[sl * CH_STRIDE + d]) = make_uint2(lo, hi);
  }
  __syncthreads();

  int w = tid >> 5, lane = tid & 31;
  // ---- K and V projection: C(64x1024) = chunk(64x512) @ [Wk|Wv]^T ----
  for (int nt = w; nt < 64; nt += 8) {
    int wrow0 = DD + nt * 16;                            // in_proj rows 512..1535
    v8f c[4] = {};                                       // 4 m-tiles (64 rows)
#pragma unroll
    for (int kk = 0; kk < 16; ++kk) {
      v16bf bf = load_fragB(Wb, wrow0, DD, kk * 32);     // one global load
#pragma unroll
      for (int mt = 0; mt < 4; ++mt)                     // 4 independent WMMAs
        c[mt] = wmma_bf16(load_fragA(chunk, mt * 16, CH_STRIDE, kk * 32), bf, c[mt]);
    }
    int nl = lane & 15;
    float bias = inb[wrow0 + nl];
    if (nt < 32) {                                       // K-projection -> LDS
      int d = nt * 16 + nl;
#pragma unroll
      for (int mt = 0; mt < 4; ++mt)
#pragma unroll
        for (int r = 0; r < 8; ++r) {
          int m = mt * 16 + r + ((lane >> 4) << 3);
          kc[m * CH_STRIDE + d] = f32_to_bf16(c[mt][r] + bias);
        }
    } else {                                             // V-projection -> ws (bf16)
      int d = (nt - 32) * 16 + nl;
#pragma unroll
      for (int mt = 0; mt < 4; ++mt)
#pragma unroll
        for (int r = 0; r < 8; ++r) {
          int m = mt * 16 + r + ((lane >> 4) << 3);
          Vb[((size_t)(b * SS + s0 + m)) * DD + d] = f32_to_bf16(c[mt][r] + bias);
        }
    }
  }
  __syncthreads();

  // ---- scores[h][16kw x 64s] = q_h(16x64) @ kc_h^T(64x64); wave = head ----
  const unsigned short* qbase = qb + (size_t)b * KW * DD;
  {
    int h = w;                                           // 8 waves = 8 heads
    v8f c[4] = {};                                       // 4 s-tiles
#pragma unroll
    for (int kk = 0; kk < 2; ++kk) {
      v16bf a = load_fragA(qbase, 0, DD, h * HD + kk * 32);  // reused 4x
#pragma unroll
      for (int st = 0; st < 4; ++st)
        c[st] = wmma_bf16(a, load_fragB(kc, st * 16, CH_STRIDE, h * HD + kk * 32), c[st]);
    }
#pragma unroll
    for (int st = 0; st < 4; ++st) {
      int sg = s0 + st * 16 + (lane & 15);
      bool pd = pad[b * SS + sg] != 0;
#pragma unroll
      for (int r = 0; r < 8; ++r) {
        int kwd = r + ((lane >> 4) << 3);
        scores[(((size_t)(b * HH + h) * KW) + kwd) * SS + sg] = pd ? c[st][r] : -1.0e9f;
      }
    }
  }
}

// ---------------------------------------------------------------------------
// Kernel S: in-place row softmax over S=2048; one block per (b,h,kw) row
// ---------------------------------------------------------------------------
__global__ __launch_bounds__(256) void k_softmax(float* __restrict__ scores) {
  __shared__ float red[9];
  int tid = threadIdx.x;
  float* p = scores + (size_t)blockIdx.x * SS;
  float v[8];
  float mx = -3.402823466e38f;
#pragma unroll
  for (int i = 0; i < 8; ++i) { v[i] = p[tid + i * 256]; mx = fmaxf(mx, v[i]); }
  mx = blockReduceMax(mx, red);
  float sum = 0.f;
#pragma unroll
  for (int i = 0; i < 8; ++i) { v[i] = __expf(v[i] - mx); sum += v[i]; }
  sum = blockReduceSum(sum, red);
  float inv = 1.0f / sum;
#pragma unroll
  for (int i = 0; i < 8; ++i) p[tid + i * 256] = v[i] * inv;
}

// ---------------------------------------------------------------------------
// Kernel P: attn_probs = renorm(mask(mean_h(weights)))  -> d_out region 3
// one block per (b,kw)
// ---------------------------------------------------------------------------
__global__ __launch_bounds__(256) void k_probs(const float* __restrict__ wts,
                                               const u8* __restrict__ kwm,
                                               const u8* __restrict__ pad,
                                               float* __restrict__ out3) {
  __shared__ float red[9];
  int b = blockIdx.x >> 4, k = blockIdx.x & 15, tid = threadIdx.x;
  bool kq = kwm[b * KW + k] != 0;
  float pv[8];
  float tot = 0.f;
#pragma unroll
  for (int i = 0; i < 8; ++i) {
    int s = tid + i * 256;
    float acc = 0.f;
    for (int h = 0; h < HH; ++h)
      acc += wts[(((size_t)(b * HH + h) * KW) + k) * SS + s];
    acc *= (1.0f / HH);
    if (!kq || pad[b * SS + s] == 0) acc = 0.f;
    pv[i] = acc;
    tot += acc;
  }
  tot = blockReduceSum(tot, red);
  float inv = 1.0f / fmaxf(tot, 1e-6f);
#pragma unroll
  for (int i = 0; i < 8; ++i)
    out3[((size_t)(b * KW + k)) * SS + tid + i * 256] = kq ? pv[i] * inv : 0.f;
}

// ---------------------------------------------------------------------------
// Kernel C: attn[b,k,h*64+d] = sum_s w[b,h,k,s] * V[b,s,h*64+d]
// one block per (b,h); LDS-staged chunks of 64 s
// ---------------------------------------------------------------------------
__global__ __launch_bounds__(256) void k_attn(const float* __restrict__ wts,
                                              const unsigned short* __restrict__ Vb,
                                              float* __restrict__ attn) {
  __shared__ float wl[KW][S_CHUNK];
  __shared__ unsigned short vl[S_CHUNK][HD];
  int b = blockIdx.x >> 3, h = blockIdx.x & 7, tid = threadIdx.x;
  int d = tid & 63, kg = tid >> 6;                       // kg in 0..3 -> 4 keywords each
  float acc[4] = {0.f, 0.f, 0.f, 0.f};
  for (int sc = 0; sc < SS / S_CHUNK; ++sc) {
    int s0 = sc * S_CHUNK;
    for (int i = 0; i < 4; ++i) {
      int idx = tid + i * 256;
      wl[idx >> 6][idx & 63] = wts[(((size_t)(b * HH + h) * KW) + (idx >> 6)) * SS + s0 + (idx & 63)];
    }
    for (int i = 0; i < 16; ++i) {
      int idx = tid + i * 256;
      vl[idx >> 6][idx & 63] = Vb[((size_t)(b * SS + s0 + (idx >> 6))) * DD + h * HD + (idx & 63)];
    }
    __syncthreads();
    for (int ss = 0; ss < S_CHUNK; ++ss) {
      float vv = bf16_to_f32(vl[ss][d]);
#pragma unroll
      for (int j = 0; j < 4; ++j) acc[j] += wl[kg * 4 + j][ss] * vv;
    }
    __syncthreads();
  }
#pragma unroll
  for (int j = 0; j < 4; ++j)
    attn[((size_t)(b * KW + kg * 4 + j)) * DD + h * HD + d] = acc[j];
}

// ---------------------------------------------------------------------------
// Kernel O: attn_out = attn @ out_w^T + out_b; keyword_context (d_out region 2);
// aggregated = masked mean over keywords -> ws.  One block per b.
// ---------------------------------------------------------------------------
__global__ __launch_bounds__(256) void k_outproj(const float* __restrict__ attn,
                                                 const float* __restrict__ ow,
                                                 const float* __restrict__ ob,
                                                 const u8* __restrict__ kwm,
                                                 float* __restrict__ out2,
                                                 float* __restrict__ agg) {
  __shared__ float al[KW * DD];                          // 32 KB
  int b = blockIdx.x, tid = threadIdx.x;
  for (int i = 0; i < 32; ++i)
    al[tid + i * 256] = attn[(size_t)b * KW * DD + tid + i * 256];
  __syncthreads();
  float cnt = 0.f;
  for (int k = 0; k < KW; ++k) cnt += (kwm[b * KW + k] != 0) ? 1.f : 0.f;
  float inv = 1.0f / fmaxf(cnt, 1.f);
  for (int nn = 0; nn < 2; ++nn) {
    int n = tid + nn * 256;
    const float* wr = ow + (size_t)n * DD;
    float a2 = 0.f;
    for (int k = 0; k < KW; ++k) {
      float o = ob[n];
      for (int d2 = 0; d2 < DD; ++d2) o += al[k * DD + d2] * wr[d2];
      float oc = (kwm[b * KW + k] != 0) ? o : 0.f;
      out2[((size_t)(b * KW + k)) * DD + n] = oc;
      a2 += oc;
    }
    agg[b * DD + n] = a2 * inv;
  }
}

// ---------------------------------------------------------------------------
// Kernel D: sdelta = tanh(rs) * (agg @ delta_w^T + delta_b); also per-b active flag
// ---------------------------------------------------------------------------
__global__ __launch_bounds__(256) void k_delta(const float* __restrict__ agg,
                                               const float* __restrict__ dw,
                                               const float* __restrict__ db,
                                               const float* __restrict__ rs,
                                               const u8* __restrict__ kwm,
                                               float* __restrict__ sdelta,
                                               float* __restrict__ actf) {
  __shared__ float ag[DD];
  int b = blockIdx.x, tid = threadIdx.x;
  ag[tid] = agg[b * DD + tid];
  ag[tid + 256] = agg[b * DD + tid + 256];
  __syncthreads();
  float scale = tanhf(rs[0]);
  for (int nn = 0; nn < 2; ++nn) {
    int n = tid + nn * 256;
    float d = db[n];
    const float* wr = dw + (size_t)n * DD;
    for (int d2 = 0; d2 < DD; ++d2) d += ag[d2] * wr[d2];
    sdelta[b * DD + n] = scale * d;
  }
  if (tid == 0) {
    int act = 0;
    for (int k = 0; k < KW; ++k) act |= (kwm[b * KW + k] != 0);
    actf[b] = act ? 1.f : 0.f;
  }
}

// ---------------------------------------------------------------------------
// Kernel L: updated = select(pad & active, LN(vision + sdelta), vision)
// one block per (b,s) row; 2 elems/thread  (memory-bound pass)
// ---------------------------------------------------------------------------
__global__ __launch_bounds__(256) void k_ln(const float* __restrict__ vision,
                                            const u8* __restrict__ pad,
                                            const float* __restrict__ sdelta,
                                            const float* __restrict__ gamma,
                                            const float* __restrict__ beta,
                                            const float* __restrict__ actf,
                                            float* __restrict__ out1) {
  __shared__ float red[9];
  int bs = blockIdx.x, b = bs >> 11, tid = threadIdx.x;
  const float* vr = vision + (size_t)bs * DD;
  float v0 = vr[tid], v1 = vr[tid + 256];
  float x0 = v0 + sdelta[b * DD + tid];
  float x1 = v1 + sdelta[b * DD + tid + 256];
  float mu = blockReduceSum(x0 + x1, red) * (1.0f / DD);
  float d0 = x0 - mu, d1 = x1 - mu;
  float var = blockReduceSum(d0 * d0 + d1 * d1, red) * (1.0f / DD);
  float inv = rsqrtf(var + 1e-5f);
  float l0 = d0 * inv * gamma[tid] + beta[tid];
  float l1 = d1 * inv * gamma[tid + 256] + beta[tid + 256];
  bool up = (pad[bs] != 0) && (actf[b] != 0.f);
  float* orow = out1 + (size_t)bs * DD;
  orow[tid]       = up ? l0 : v0;
  orow[tid + 256] = up ? l1 : v1;
}

// ---------------------------------------------------------------------------
// Launcher
// ---------------------------------------------------------------------------
extern "C" void kernel_launch(void* const* d_in, const int* in_sizes, int n_in,
                              void* d_out, int out_size, void* d_ws, size_t ws_size,
                              hipStream_t stream) {
  const float* vision = (const float*)d_in[0];
  const u8*    pad    = (const u8*)d_in[1];               // jnp bool -> 1 byte
  const float* kq     = (const float*)d_in[2];
  const u8*    kwm    = (const u8*)d_in[3];
  const float* ipw    = (const float*)d_in[4];
  const float* ipb    = (const float*)d_in[5];
  const float* ow     = (const float*)d_in[6];
  const float* ob     = (const float*)d_in[7];
  const float* dw     = (const float*)d_in[8];
  const float* db     = (const float*)d_in[9];
  const float* gamma  = (const float*)d_in[10];
  const float* beta   = (const float*)d_in[11];
  const float* rs     = (const float*)d_in[12];

  float* out1 = (float*)d_out;                                       // updated
  float* out2 = out1 + (size_t)BB * SS * DD;                         // keyword_context
  float* out3 = out2 + (size_t)BB * KW * DD;                         // attn_probs

  // workspace layout (bytes, 256-aligned)
  char* ws = (char*)d_ws;
  size_t o = 0;
  unsigned short* Wb = (unsigned short*)(ws + o); o += (size_t)3 * DD * DD * 2;        // 1.5 MB
  unsigned short* qb = (unsigned short*)(ws + o); o += (size_t)BB * KW * DD * 2;       // 0.5 MB
  unsigned short* Vb = (unsigned short*)(ws + o); o += (size_t)BB * SS * DD * 2;       // 64 MB
  float* scores = (float*)(ws + o); o += (size_t)BB * HH * KW * SS * 4;                // 32 MB
  float* attn   = (float*)(ws + o); o += (size_t)BB * KW * DD * 4;                     // 1 MB
  float* agg    = (float*)(ws + o); o += (size_t)BB * DD * 4;
  float* sdelta = (float*)(ws + o); o += (size_t)BB * DD * 4;
  float* actf   = (float*)(ws + o); o += 256;
  (void)ws_size; (void)n_in; (void)in_sizes; (void)out_size;

  // 1) weights -> bf16
  k_convW<<<(3 * DD * DD / 4) / 256, 256, 0, stream>>>(ipw, Wb);
  // 2) q projection (WMMA), scale folded
  k_qproj<<<BB, 256, 0, stream>>>(kq, Wb, ipb, qb);
  // 3) fused K/V projection + scores (dominant WMMA kernel)
  size_t lds = (size_t)2 * S_CHUNK * CH_STRIDE * sizeof(unsigned short);  // 133120 B
  k_kv_scores<<<BB * (SS / S_CHUNK), 256, lds, stream>>>(vision, pad, Wb, ipb, qb, Vb, scores);
  // 4) softmax rows (in place)
  k_softmax<<<BB * HH * KW, 256, 0, stream>>>(scores);
  // 5) attn_probs output
  k_probs<<<BB * KW, 256, 0, stream>>>(scores, kwm, pad, out3);
  // 6) attn = weights @ V
  k_attn<<<BB * HH, 256, 0, stream>>>(scores, Vb, attn);
  // 7) out projection + keyword_context + aggregation
  k_outproj<<<BB, 256, 0, stream>>>(attn, ow, ob, kwm, out2, agg);
  // 8) delta + tanh scale + active flags
  k_delta<<<BB, 256, 0, stream>>>(agg, dw, db, rs, kwm, sdelta, actf);
  // 9) residual + LayerNorm + mask select
  k_ln<<<BB * SS, 256, 0, stream>>>(vision, pad, sdelta, gamma, beta, actf, out1);
}
